// Gate_45019847197030
// MI455X (gfx1250) — compile-verified, matrix operand order checked by reading, not run on hardware
//
#include <hip/hip_runtime.h>
#include <hip/hip_bf16.h>
#include <stdint.h>

// ---- CDNA5 WMMA types ----
typedef __attribute__((ext_vector_type(16))) __bf16 v16bf;
typedef __attribute__((ext_vector_type(8)))  float  v8f;

#define T_TOK 65536
#define DK    2048
#define EK    128
#define WAVES 4
#define BLK_T (WAVES * 16)   // 64 tokens per block
#define KCH   32             // K per WMMA step (bf16)

union BFrag {
    unsigned int u[8];
    v16bf        v;
};

// f32 pair -> packed bf16 dword, RNE, using CDNA5 V_FMA_MIX*_BF16 (2 VALU ops).
// (ISA 15.10 opcodes 62/63: D[15:0]/[31:16].bf16 = cvt(fma(f32,1.0,0)))
__device__ __forceinline__ unsigned int pk_bf16(float lo, float hi) {
    unsigned int d;
    asm("v_fma_mixlo_bf16 %0, %1, 1.0, 0\n\t"
        "v_fma_mixhi_bf16 %0, %2, 1.0, 0"
        : "=&v"(d)
        : "v"(lo), "v"(hi));
    return d;
}

// convert 8 consecutive f32 (two float4) into 4 packed-bf16 dwords
__device__ __forceinline__ uint4 pk_bf16_x8(const float4 f0, const float4 f1) {
    uint4 r;
    r.x = pk_bf16(f0.x, f0.y);
    r.y = pk_bf16(f0.z, f0.w);
    r.z = pk_bf16(f1.x, f1.y);
    r.w = pk_bf16(f1.z, f1.w);
    return r;
}

__global__ __launch_bounds__(WAVES * 32)
void moe_router_kernel(const float* __restrict__ x,
                       const float* __restrict__ W,
                       const float* __restrict__ bias,
                       float* __restrict__ out) {
    // Double-buffered B fragments, pre-swizzled into WMMA lane layout:
    // ldsB[buf][ntile][lane][vgpr] (bf16 pair per dword) = 2 x 8 KB
    __shared__ unsigned int ldsB[2][8][32][8];
    // per-wave logits tile, padded stride 129 (conflict-free column reads)
    __shared__ float ldsL[WAVES][16][129];

    const int tid  = threadIdx.x;
    const int wave = tid >> 5;
    const int lane = tid & 31;
    const int lm   = lane & 15;   // M (or N) index within 16
    const int lh   = lane >> 4;   // half-lane selects K sub-range

    const size_t tb = (size_t)blockIdx.x * BLK_T;  // block token base
    const size_t mt = tb + (size_t)wave * 16;      // wave token base

    const float* xrow = x + (mt + lm) * (size_t)DK;

    v8f zero = {};
    v8f acc[8];
#pragma unroll
    for (int i = 0; i < 8; ++i) acc[i] = zero;

    // cooperative W-tile fill into buffer p for K-chunk kc:
    // 512 tasks (8 nt x 32 lane-slots x 2 vgpr-groups), 128 threads -> 4 each,
    // each task: 8 consecutive K f32 -> 4 packed dwords -> one ds_store_b128
    auto fill_wtile = [&](int p, int kc) {
#pragma unroll
        for (int j = 0; j < 4; ++j) {
            int q  = tid + j * 128;        // 0..511
            int nt = q >> 6;               // n-tile 0..7
            int l  = (q >> 1) & 31;        // lane slot
            int vg = q & 1;                // vgpr group (0: v0..3, 1: v4..7)
            int e  = nt * 16 + (l & 15);   // expert (B column)
            int h  = l >> 4;
            int k  = kc + (vg << 4) + (h << 3);   // 8 consecutive K
            const float4* ptr = (const float4*)(W + (size_t)e * DK + k);
            uint4 pk = pk_bf16_x8(ptr[0], ptr[1]);
            *(uint4*)&ldsB[p][nt][l][vg << 2] = pk;
        }
    };

    // A fragment from global x (two b128 loads per vgpr group) + 8 WMMAs
    auto do_chunk = [&](int p, int kc, bool prefetch_next) {
        BFrag a;
        const float* xr = xrow + kc;
#pragma unroll
        for (int vg = 0; vg < 2; ++vg) {
            int k = (vg << 4) + (lh << 3);     // 8 consecutive K
            const float4* ptr = (const float4*)(xr + k);
            uint4 pk = pk_bf16_x8(ptr[0], ptr[1]);
            a.u[vg * 4 + 0] = pk.x;
            a.u[vg * 4 + 1] = pk.y;
            a.u[vg * 4 + 2] = pk.z;
            a.u[vg * 4 + 3] = pk.w;
        }
        if (prefetch_next) __builtin_prefetch(xr + KCH, 0, 0);

#pragma unroll
        for (int nt = 0; nt < 8; ++nt) {
            BFrag b;
#pragma unroll
            for (int v = 0; v < 8; ++v) b.u[v] = ldsB[p][nt][lane][v];
            acc[nt] = __builtin_amdgcn_wmma_f32_16x16x32_bf16(
                false, a.v, false, b.v, (short)0, acc[nt], false, false);
        }
    };

    fill_wtile(0, 0);
    __syncthreads();

    // steady state: straight-line body, one barrier per chunk
    int p = 0;
    for (int kc = 0; kc < DK - KCH; kc += KCH) {
        fill_wtile(p ^ 1, kc + KCH);   // stage next chunk into other buffer
        do_chunk(p, kc, true);         // compute current chunk
        __syncthreads();
        p ^= 1;
    }
    // peeled last chunk (no fill, no prefetch, no barrier)
    do_chunk(p, DK - KCH, false);

    // ---- spill logits tile to LDS: ldsL[wave][token m][expert n] ----
    __syncthreads();
#pragma unroll
    for (int nt = 0; nt < 8; ++nt) {
#pragma unroll
        for (int i = 0; i < 8; ++i) {
            int m = i + 8 * lh;
            int n = nt * 16 + lm;
            ldsL[wave][m][n] = acc[nt][i];
        }
    }
    __syncthreads();

    // ---- per-token routing: one lane per token (lanes 0..15 of each wave) ----
    if (lane < 16) {
        const size_t tok = mt + lane;
        float* Lrow = &ldsL[wave][lane][0];

        // pass 1: max logit (with bias) for a stable exp
        float mx = -3.4e38f;
#pragma unroll 4
        for (int e = 0; e < EK; ++e) {
            float v = Lrow[e] + bias[e];
            mx = fmaxf(mx, v);
        }

        // pass 2: scores = exp(l-mx); track per-group top-2 (softmax Z cancels)
        float g1a = -1.0f, g2a = -1.0f, g1b = -1.0f, g2b = -1.0f;
#pragma unroll 4
        for (int e = 0; e < EK; ++e) {
            float s = __expf(Lrow[e] + bias[e] - mx);
            Lrow[e] = s;
            if (e < 64) {
                if (s > g1a) { g2a = g1a; g1a = s; }
                else if (s > g2a) { g2a = s; }
            } else {
                if (s > g1b) { g2b = g1b; g1b = s; }
                else if (s > g2b) { g2b = s; }
            }
        }
        const int base = ((g1a + g2a) >= (g1b + g2b)) ? 0 : 64;

        // pass 3: top-4 within selected group (strict '>' => lowest index wins ties)
        float v0 = -1.f, v1 = -1.f, v2 = -1.f, v3 = -1.f;
        int   i0 = 0,    i1 = 0,    i2 = 0,    i3 = 0;
#pragma unroll 4
        for (int e = 0; e < 64; ++e) {
            float s = Lrow[base + e];
            if (s > v0)      { v3=v2; i3=i2; v2=v1; i2=i1; v1=v0; i1=i0; v0=s; i0=e; }
            else if (s > v1) { v3=v2; i3=i2; v2=v1; i2=i1; v1=s;  i1=e; }
            else if (s > v2) { v3=v2; i3=i2; v2=s;  i2=e; }
            else if (s > v3) { v3=s;  i3=e; }
        }

        float ws  = fmaxf(v0 + v1 + v2 + v3, 1e-9f);
        float inv = 1.0f / ws;

        float* ow = out + tok * 4;
        int*   oi = (int*)(out + (size_t)T_TOK * 4) + tok * 4;
        ow[0] = v0 * inv; ow[1] = v1 * inv; ow[2] = v2 * inv; ow[3] = v3 * inv;
        oi[0] = base + i0; oi[1] = base + i1; oi[2] = base + i2; oi[3] = base + i3;
    }
}

extern "C" void kernel_launch(void* const* d_in, const int* in_sizes, int n_in,
                              void* d_out, int out_size, void* d_ws, size_t ws_size,
                              hipStream_t stream) {
    const float* x    = (const float*)d_in[0];
    const float* W    = (const float*)d_in[1];
    const float* bias = (const float*)d_in[2];
    float* out        = (float*)d_out;

    dim3 grid(T_TOK / BLK_T);   // 1024 blocks
    dim3 block(WAVES * 32);     // 128 threads = 4 wave32s

    hipLaunchKernelGGL(moe_router_kernel, grid, block, 0, stream, x, W, bias, out);
}
